// UserModelPW_87376814670341
// MI455X (gfx1250) — compile-verified, matrix operand order checked by reading, not run on hardware
//
#include <hip/hip_runtime.h>

#define S_SEC   1024
#define F_FEAT  3952
#define KP      3968      /* padded K = 124 * 32 */
#define KSTEPS  124
#define PW_N    4
#define NDISP   8192

typedef __attribute__((ext_vector_type(16))) __bf16        v16bf;
typedef __attribute__((ext_vector_type(8)))  float         v8f;
typedef __attribute__((ext_vector_type(4)))  float         f32x4;
typedef __attribute__((ext_vector_type(4)))  unsigned int  u32x4;
typedef __attribute__((ext_vector_type(4)))  int           i32x4;

typedef __attribute__((address_space(1))) i32x4* gas_i32x4;   // global
typedef __attribute__((address_space(3))) i32x4* las_i32x4;   // LDS

// ---- feature probes (compile-safe fallbacks) -------------------------------
#if defined(__has_builtin)
#  if __has_builtin(__builtin_amdgcn_global_load_async_to_lds_b128) && \
      __has_builtin(__builtin_amdgcn_s_wait_asynccnt)
#    define USE_ASYNC_LDS 1
#  else
#    define USE_ASYNC_LDS 0
#  endif
#  if __has_builtin(__builtin_amdgcn_sched_group_barrier)
#    define HAVE_SCHED_GROUP 1
#  else
#    define HAVE_SCHED_GROUP 0
#  endif
#else
#  define USE_ASYNC_LDS 0
#  define HAVE_SCHED_GROUP 0
#endif

struct U256 { u32x4 a, b; };

static __device__ inline v16bf frag_from(u32x4 lo, u32x4 hi) {
  U256 t; t.a = lo; t.b = hi;
  return __builtin_bit_cast(v16bf, t);
}

static __device__ inline unsigned short f2bf(float x) {
  __bf16 h = (__bf16)x;
  return __builtin_bit_cast(unsigned short, h);
}

// ---------------------------------------------------------------------------
// Pack W1 sub-matrices into bf16, pre-swizzled into the CDNA5 16-bit
// B-fragment layout: fragment block (t, nb) holds 32x16, lane L keeps col
// n = nb*16 + (L&15); element idx e<8 -> K = t*32 + ((L>>4)<<3) + e,
// e>=8 -> K = ... + e + 8  (K 0..7 / 16..23 lanes<16, 8..15 / 24..31 lanes>=16).
// mode 0: W1a packed (N=256, col = p*64+j  ->  W1[p*F + k, j])
// mode 1: W1b        (N=64,                ->  W1[4*F + k, n])
// K padded with zeros up to KP.
// ---------------------------------------------------------------------------
__global__ void pack_w1(const float* __restrict__ W1, unsigned short* __restrict__ out,
                        int N, int mode) {
  int tid = blockIdx.x * blockDim.x + threadIdx.x;
  int total = KSTEPS * (N >> 4) * 512;
  if (tid >= total) return;
  int fb   = tid >> 9;
  int r    = tid & 511;
  int lane = r >> 4;
  int idx  = r & 15;
  int nbg  = fb % (N >> 4);
  int t    = fb / (N >> 4);
  int k    = t * 32 + ((lane >> 4) << 3) + (idx < 8 ? idx : idx + 8);
  int n    = (nbg << 4) + (lane & 15);
  float v = 0.f;
  if (k < F_FEAT) {
    if (mode == 0) {
      int p = n >> 6, j = n & 63;
      v = W1[((size_t)(p * F_FEAT + k)) * 64 + j];
    } else {
      v = W1[((size_t)(PW_N * F_FEAT + k)) * 64 + n];
    }
  }
  out[tid] = f2bf(v);
}

// ---------------------------------------------------------------------------
// C[M,N] = bf16(A_fp32[M x 3952]) @ Bpk[KP x N]   (N multiple of 64)
// Block: 256 threads = 8 waves; wave w owns rows blockIdx.x*128 + w*16 .. +15,
// cols blockIdx.y*64 .. +63 as 4 WMMA accumulators. B tile (32 x 64 bf16 =
// 4 KB) staged through double-buffered LDS in fragment order, preferring the
// CDNA5 async global->LDS path when available.
// ---------------------------------------------------------------------------
__global__ __launch_bounds__(256) void gemm_a32_bw16(
    const float* __restrict__ A, const u32x4* __restrict__ Bpk,
    float* __restrict__ C, int lda, int N, int ldc)
{
  __shared__ u32x4 ldsB[2][256];

  const int wave = threadIdx.x >> 5;
  const int lane = threadIdx.x & 31;
  const int row  = blockIdx.x * 128 + wave * 16 + (lane & 15);
  const int koff = (lane >> 4) << 3;
  const float* arow = A + (size_t)row * lda;
  const int nb16   = N >> 4;
  const int nbBase = blockIdx.y * 4;

#if USE_ASYNC_LDS
  auto fill = [&](int t) {
    int c      = threadIdx.x;          // 256 chunks of 16B = 4KB tile
    int nb_l   = c >> 6;
    int within = c & 63;
    const u32x4* src = &Bpk[((size_t)(t * nb16 + nbBase + nb_l)) * 64 + within];
    __builtin_amdgcn_global_load_async_to_lds_b128(
        (gas_i32x4)src,
        (las_i32x4)&ldsB[t & 1][c],
        0, 0);
  };
#else
  auto fill = [&](int t) {
    int c      = threadIdx.x;          // 256 chunks of 16B = 4KB tile
    int nb_l   = c >> 6;
    int within = c & 63;
    ldsB[t & 1][c] = Bpk[((size_t)(t * nb16 + nbBase + nb_l)) * 64 + within];
  };
#endif

  v8f acc0 = {}, acc1 = {}, acc2 = {}, acc3 = {};
  fill(0);
  for (int t = 0; t < KSTEPS; ++t) {
#if USE_ASYNC_LDS
    __builtin_amdgcn_s_wait_asynccnt(0);   // own async fills landed in LDS
#endif
    __syncthreads();                       // tile t published to all waves
    if (t + 1 < KSTEPS) fill(t + 1);

    const int kb = t * 32 + koff;
    f32x4 a0 = *(const f32x4*)(arow + kb);
    f32x4 a1 = *(const f32x4*)(arow + kb + 4);
    v16bf av;
    av[0] = (__bf16)a0.x; av[1] = (__bf16)a0.y; av[2] = (__bf16)a0.z; av[3] = (__bf16)a0.w;
    av[4] = (__bf16)a1.x; av[5] = (__bf16)a1.y; av[6] = (__bf16)a1.z; av[7] = (__bf16)a1.w;
    if (t < KSTEPS - 1) {
      f32x4 a2 = *(const f32x4*)(arow + kb + 16);
      f32x4 a3 = *(const f32x4*)(arow + kb + 20);
      av[8]  = (__bf16)a2.x; av[9]  = (__bf16)a2.y; av[10] = (__bf16)a2.z; av[11] = (__bf16)a2.w;
      av[12] = (__bf16)a3.x; av[13] = (__bf16)a3.y; av[14] = (__bf16)a3.z; av[15] = (__bf16)a3.w;
    } else {
      // K tail: 3952..3967 are zero-padded in B; keep A contribution zero too.
      __bf16 z = (__bf16)0.f;
      av[8] = z; av[9] = z; av[10] = z; av[11] = z;
      av[12] = z; av[13] = z; av[14] = z; av[15] = z;
    }
    if (t + 4 < KSTEPS) __builtin_prefetch(arow + kb + 128, 0, 3);

    const int lb = t & 1;
    const int lo = lane * 2;
    v16bf b0 = frag_from(ldsB[lb][lo],       ldsB[lb][lo + 1]);
    v16bf b1 = frag_from(ldsB[lb][64 + lo],  ldsB[lb][64 + lo + 1]);
    v16bf b2 = frag_from(ldsB[lb][128 + lo], ldsB[lb][128 + lo + 1]);
    v16bf b3 = frag_from(ldsB[lb][192 + lo], ldsB[lb][192 + lo + 1]);

    acc0 = __builtin_amdgcn_wmma_f32_16x16x32_bf16(false, av, false, b0, (short)0, acc0, false, false);
    acc1 = __builtin_amdgcn_wmma_f32_16x16x32_bf16(false, av, false, b1, (short)0, acc1, false, false);
    acc2 = __builtin_amdgcn_wmma_f32_16x16x32_bf16(false, av, false, b2, (short)0, acc2, false, false);
    acc3 = __builtin_amdgcn_wmma_f32_16x16x32_bf16(false, av, false, b3, (short)0, acc3, false, false);

#if HAVE_SCHED_GROUP
    // Group the 8 DS fragment reads ahead of the 4 WMMAs so the matrix ops
    // issue back-to-back behind a single s_wait_dscnt.
    __builtin_amdgcn_sched_group_barrier(0x020, 4, 0);  // VMEM reads (A)
    __builtin_amdgcn_sched_group_barrier(0x002, 8, 0);  // VALU cvts
    __builtin_amdgcn_sched_group_barrier(0x100, 8, 0);  // DS reads (B frags)
    __builtin_amdgcn_sched_group_barrier(0x008, 4, 0);  // WMMA
#endif
  }

  // C/D layout: VGPR r -> row M = r (lanes 0-15) or r+8 (lanes 16-31), col = lane&15.
  const int colBase = blockIdx.y * 64 + (lane & 15);
  const int rowBase = blockIdx.x * 128 + wave * 16 + ((lane >> 4) << 3);
#pragma unroll
  for (int r = 0; r < 8; ++r) {
    float* crow = C + (size_t)(rowBase + r) * ldc + colBase;
    crow[0]  = acc0[r];
    crow[16] = acc1[r];
    crow[32] = acc2[r];
    crow[48] = acc3[r];
  }
}

// ---------------------------------------------------------------------------
// Ahist[i, j] = sum_p sum_{d<=min(i,19)} pw[p,d] * Y[i-d, p*64 + j]
// (banded conv applied AFTER the GEMM; equals concat_history @ W1a)
// ---------------------------------------------------------------------------
__global__ void hist_conv(const float* __restrict__ Y, const float* __restrict__ pw,
                          float* __restrict__ Ahist) {
  int tid = blockIdx.x * blockDim.x + threadIdx.x;   // S_SEC * 64 threads
  int i = tid >> 6, j = tid & 63;
  int dmax = (i < 19) ? i : 19;
  float acc = 0.f;
#pragma unroll
  for (int p = 0; p < 4; ++p) {
    const float* yp  = Y + p * 64 + j;
    const float* pwp = pw + p * 20;
    for (int d = 0; d <= dmax; ++d)
      acc += pwp[d] * yp[(size_t)(i - d) * 256];
  }
  Ahist[tid] = acc;
}

// ---------------------------------------------------------------------------
// Per displayed item: h1 = elu(Bdisp + Ahist[sec] + b1); h2 = elu(h1@W2+b2);
// u = h2@W3 + b3; also exp(u).
// ---------------------------------------------------------------------------
__global__ __launch_bounds__(64) void mlp_kernel(
    const float* __restrict__ Bdisp, const float* __restrict__ Ahist,
    const int* __restrict__ sec_ind,
    const float* __restrict__ b1, const float* __restrict__ W2,
    const float* __restrict__ b2, const float* __restrict__ W3,
    const float* __restrict__ b3,
    float* __restrict__ u, float* __restrict__ eu)
{
  __shared__ float sh1[64];
  __shared__ float sh2[64];
  const int dd = blockIdx.x;
  const int j  = threadIdx.x;
  const int sec = sec_ind[dd];

  float v = Bdisp[(size_t)dd * 64 + j] + Ahist[(size_t)sec * 64 + j] + b1[j];
  sh1[j] = (v > 0.f) ? v : (expf(v) - 1.f);
  __syncthreads();

  float a = b2[j];
#pragma unroll 8
  for (int k = 0; k < 64; ++k) a += sh1[k] * W2[k * 64 + j];
  sh2[j] = (a > 0.f) ? a : (expf(a) - 1.f);
  __syncthreads();

  sh1[j] = sh2[j] * W3[j];
  __syncthreads();
  if (j == 0) {
    float s = b3[0];
    for (int k = 0; k < 64; ++k) s += sh1[k];
    u[dd]  = s;
    eu[dd] = expf(s);
  }
}

// ---------------------------------------------------------------------------
// Per section: softmax denominator over its 8 items, per-section loss term,
// and argmax / top-2 of the (sparse, duplicate-combined) dense_exp row.
// ---------------------------------------------------------------------------
__global__ void section_kernel(const float* __restrict__ u, const float* __restrict__ eu,
                               const int* __restrict__ disp_idx,
                               const int* __restrict__ click_idx,
                               const int* __restrict__ click_sub,
                               float* __restrict__ lossv,
                               float* __restrict__ p1v, float* __restrict__ p2v)
{
  int i = blockIdx.x * blockDim.x + threadIdx.x;
  if (i >= S_SEC) return;
  int cols[8]; float ev[8];
  float sum = 0.f;
#pragma unroll
  for (int q = 0; q < 8; ++q) {
    cols[q] = disp_idx[(i * 8 + q) * 2 + 1];
    ev[q]   = eu[i * 8 + q];
    sum    += ev[q];
  }
  float best = -1.f, second = -1.f;
  int bcol = -1, scol = -1;
#pragma unroll
  for (int q = 0; q < 8; ++q) {
    bool first = true;
    for (int r = 0; r < q; ++r) if (cols[r] == cols[q]) { first = false; break; }
    if (!first) continue;
    float tot = ev[q];
    for (int r = q + 1; r < 8; ++r) if (cols[r] == cols[q]) tot += ev[r];
    if (tot > best)        { second = best; scol = bcol; best = tot; bcol = cols[q]; }
    else if (tot > second) { second = tot; scol = cols[q]; }
  }
  int ac = click_idx[i * 2 + 1];
  p1v[i] = (ac == bcol) ? 1.f : 0.f;
  p2v[i] = (ac == bcol || ac == scol) ? 1.f : 0.f;
  lossv[i] = -u[click_sub[i]] + logf(sum + 1.f);
}

// ---------------------------------------------------------------------------
// Deterministic tree reduction -> 7 outputs.
// ---------------------------------------------------------------------------
__global__ void reduce_out(const float* __restrict__ lossv,
                           const float* __restrict__ p1v,
                           const float* __restrict__ p2v,
                           float* __restrict__ out)
{
  __shared__ float sl[256], s1[256], s2[256];
  int t = threadIdx.x;
  float a = 0.f, b = 0.f, c = 0.f;
  for (int i = t; i < S_SEC; i += 256) { a += lossv[i]; b += p1v[i]; c += p2v[i]; }
  sl[t] = a; s1[t] = b; s2[t] = c;
  __syncthreads();
  for (int s = 128; s > 0; s >>= 1) {
    if (t < s) { sl[t] += sl[t + s]; s1[t] += s1[t + s]; s2[t] += s2[t + s]; }
    __syncthreads();
  }
  if (t == 0) {
    float ec = (float)S_SEC;   // click scatter adds exactly one 1.0 per row
    out[0] = sl[0] / ec;
    out[1] = s1[0] / ec;
    out[2] = s2[0] / ec;
    out[3] = sl[0];
    out[4] = s1[0];
    out[5] = s2[0];
    out[6] = ec;
  }
}

// ---------------------------------------------------------------------------
extern "C" void kernel_launch(void* const* d_in, const int* in_sizes, int n_in,
                              void* d_out, int out_size, void* d_ws, size_t ws_size,
                              hipStream_t stream)
{
  (void)in_sizes; (void)n_in; (void)out_size; (void)ws_size;
  const float* disp = (const float*)d_in[0];   // (8192, 3952)
  const float* Xs   = (const float*)d_in[1];   // (1024, 3952)
  const float* pw   = (const float*)d_in[2];   // (4, 20)
  const float* W1   = (const float*)d_in[3];   // (19760, 64)
  const float* b1   = (const float*)d_in[4];
  const float* W2   = (const float*)d_in[5];
  const float* b2   = (const float*)d_in[6];
  const float* W3   = (const float*)d_in[7];
  const float* b3   = (const float*)d_in[8];
  const int* click_idx = (const int*)d_in[9];   // (1024, 2)
  const int* disp_idx  = (const int*)d_in[10];  // (8192, 2)
  const int* sec_ind   = (const int*)d_in[11];  // (8192,)
  const int* click_sub = (const int*)d_in[14];  // (1024,)

  char* ws = (char*)d_ws;
  size_t off = 0;
  auto alloc = [&](size_t bytes) -> void* {
    off = (off + 255) & ~(size_t)255;
    void* p = ws + off;
    off += bytes;
    return p;
  };
  unsigned short* Bpk1 = (unsigned short*)alloc((size_t)KP * 256 * 2);
  unsigned short* Bpk2 = (unsigned short*)alloc((size_t)KP * 64 * 2);
  float* Y     = (float*)alloc((size_t)S_SEC * 256 * 4);
  float* Bdisp = (float*)alloc((size_t)NDISP * 64 * 4);
  float* Ahist = (float*)alloc((size_t)S_SEC * 64 * 4);
  float* u     = (float*)alloc((size_t)NDISP * 4);
  float* eu    = (float*)alloc((size_t)NDISP * 4);
  float* lossv = (float*)alloc((size_t)S_SEC * 4);
  float* p1v   = (float*)alloc((size_t)S_SEC * 4);
  float* p2v   = (float*)alloc((size_t)S_SEC * 4);

  { int total = KSTEPS * 16 * 512;
    pack_w1<<<(total + 255) / 256, 256, 0, stream>>>(W1, Bpk1, 256, 0); }
  { int total = KSTEPS * 4 * 512;
    pack_w1<<<(total + 255) / 256, 256, 0, stream>>>(W1, Bpk2, 64, 1); }

  // G1: Y(1024,256) = Xs @ W1a_packed
  gemm_a32_bw16<<<dim3(S_SEC / 128, 256 / 64), 256, 0, stream>>>(
      Xs, (const u32x4*)Bpk1, Y, F_FEAT, 256, 256);
  // G2: Bdisp(8192,64) = disp @ W1b
  gemm_a32_bw16<<<dim3(NDISP / 128, 1), 256, 0, stream>>>(
      disp, (const u32x4*)Bpk2, Bdisp, F_FEAT, 64, 64);

  hist_conv<<<(S_SEC * 64) / 256, 256, 0, stream>>>(Y, pw, Ahist);
  mlp_kernel<<<NDISP, 64, 0, stream>>>(Bdisp, Ahist, sec_ind, b1, W2, b2, W3, b3, u, eu);
  section_kernel<<<(S_SEC + 255) / 256, 256, 0, stream>>>(
      u, eu, disp_idx, click_idx, click_sub, lossv, p1v, p2v);
  reduce_out<<<1, 256, 0, stream>>>(lossv, p1v, p2v, (float*)d_out);
}
